// CrossSourceMHA_43954695308073
// MI455X (gfx1250) — matrix operand order repaired and checked by reading, not compiled, not run on hardware
//
#include <hip/hip_runtime.h>
#include <hip/hip_bf16.h>
#include <stdint.h>

typedef _Float16 f16;
typedef __attribute__((ext_vector_type(16))) _Float16 v16h;
typedef __attribute__((ext_vector_type(8)))  float    v8f;

#define B_      8
#define C_      128
#define LQ_     4096
#define LK_     1024
#define HEADS_  4
#define DH_     32
#define EPS_    1e-5f
#define SCALE_  0.17677669529663687f   /* (C/HEADS)^-0.5 = 32^-0.5 */

// ---------------------------------------------------------------------------
// WMMA helpers (CDNA5 gfx1250, wave32).  D = A(16x32 f16) * B(32x16 f16) + C.
// A layout: lane 0-15 -> M=lane, dword i holds K pairs {0,2,...,14} (K=0..15);
//           lane 16-31 -> M=lane-16, K pairs {16,18,...,30} (K=16..31).
// B layout: lane gives N; lanes 0-15 hold K=0..15 (2/dword), lanes 16-31 K=16..31.
// C/D: lane 0-15 -> N=lane, vgpr r -> M=r ; lane 16-31 -> M=r+8.
// ---------------------------------------------------------------------------
__device__ inline v8f zero8() {
  v8f z;
#pragma unroll
  for (int i = 0; i < 8; ++i) z[i] = 0.0f;
  return z;
}

__device__ inline v16h load_a16(const f16* __restrict__ base, int ld, int m0, int k0, int lane) {
  union { v16h v; uint32_t u[8]; } f;
  const int half = lane >> 4;
  const int m = m0 + (lane & 15);
  const f16* row = base + (size_t)m * ld + k0 + half * 16;
  const uint32_t* r0 = (const uint32_t*)row;         // K = half*16 + {0,2,4,6}
  const uint32_t* r1 = (const uint32_t*)(row + 8);   // K = half*16 + {8,10,12,14}
#pragma unroll
  for (int i = 0; i < 4; ++i) f.u[i] = r0[i];
#pragma unroll
  for (int i = 0; i < 4; ++i) f.u[4 + i] = r1[i];
  return f.v;
}

// B stored n-major: buf[n][k] with row stride ldk (elements); contiguous 32B per lane.
__device__ inline v16h load_b16(const f16* __restrict__ base, int ldk, int n0, int k0, int lane) {
  union { v16h v; uint32_t u[8]; } f;
  const int half = lane >> 4;
  const int n = n0 + (lane & 15);
  const uint32_t* r0 = (const uint32_t*)(base + (size_t)n * ldk + k0 + half * 16);
#pragma unroll
  for (int i = 0; i < 8; ++i) f.u[i] = r0[i];
  return f.v;
}

__device__ inline v8f wmma_f16(v16h a, v16h b, v8f c) {
  return __builtin_amdgcn_wmma_f32_16x16x32_f16(false, a, false, b, (short)0, c, false, false);
}

// CDNA5 async global->LDS copy of 16 bytes (ASYNCcnt-tracked, no VGPR data).
__device__ inline void async_copy_b128(uint32_t lds_byte_addr, uint64_t global_addr) {
  asm volatile("global_load_async_to_lds_b128 %0, %1, off"
               :: "v"(lds_byte_addr), "v"(global_addr) : "memory");
}
__device__ inline void wait_async0() {
  asm volatile("s_wait_asynccnt 0x0" ::: "memory");
}
__device__ inline uint32_t lds_addr32(const void* p) {
  return (uint32_t)(uintptr_t)p;  // LDS aperture: addr[31:0] is the LDS offset
}

// ---------------------------------------------------------------------------
// K1: avg-pool k/v, per-(path,b,c) sum & sumsq.  path0=q(4096px) path1=k path2=v.
// grid = 3*B*C blocks of 128 threads.
// ---------------------------------------------------------------------------
__global__ void pool_stats_kernel(const float* __restrict__ q_src,
                                  const float* __restrict__ k_src,
                                  const float* __restrict__ v_src,
                                  float* __restrict__ pooled_k,
                                  float* __restrict__ pooled_v,
                                  float* __restrict__ stats /* [3*B*C][2] */) {
  const int idx  = blockIdx.x;
  const int path = idx / (B_ * C_);
  const int bc   = idx % (B_ * C_);
  const int t    = threadIdx.x;
  float s = 0.f, s2 = 0.f;
  if (path == 0) {
    const float* row = q_src + (size_t)bc * LQ_;
    for (int i = t; i < LQ_; i += 128) { float x = row[i]; s += x; s2 += x * x; }
  } else {
    const float* src = (path == 1 ? k_src : v_src) + (size_t)bc * LQ_;
    float*       dst = (path == 1 ? pooled_k : pooled_v) + (size_t)bc * LK_;
    for (int i = t; i < LK_; i += 128) {
      int py = i >> 5, px = i & 31;
      const float* p0 = src + (py * 2) * 64 + px * 2;
      float x = 0.25f * (p0[0] + p0[1] + p0[64] + p0[65]);
      dst[i] = x; s += x; s2 += x * x;
    }
  }
  __shared__ float rs[128], rq[128];
  rs[t] = s; rq[t] = s2;
  __syncthreads();
  for (int o = 64; o > 0; o >>= 1) {
    if (t < o) { rs[t] += rs[t + o]; rq[t] += rq[t + o]; }
    __syncthreads();
  }
  if (t == 0) { stats[idx * 2 + 0] = rs[0]; stats[idx * 2 + 1] = rq[0]; }
}

// ---------------------------------------------------------------------------
// K2: fold GN (or GN∘GN) + SCALE into conv weights.  W'[o][c] f16, b'[o] f32.
// grid = 3*B + 1 (last block converts Wp).  128 threads (thread = channel).
// ---------------------------------------------------------------------------
__global__ void fold_kernel(const float* __restrict__ Wq, const float* __restrict__ bq,
                            const float* __restrict__ Wk, const float* __restrict__ bk,
                            const float* __restrict__ Wv, const float* __restrict__ bv,
                            const float* __restrict__ Wp, const float* __restrict__ bp,
                            const float* __restrict__ g_nq, const float* __restrict__ b_nq,
                            const float* __restrict__ g_nk, const float* __restrict__ b_nk,
                            const float* __restrict__ g_nv, const float* __restrict__ b_nv,
                            const float* __restrict__ g_srk, const float* __restrict__ b_srk,
                            const float* __restrict__ g_srv, const float* __restrict__ b_srv,
                            const float* __restrict__ stats,
                            f16* __restrict__ Wfold, float* __restrict__ bfold) {
  const int bid = blockIdx.x;
  const int t   = threadIdx.x;  // channel (pass 1), out-channel (pass 2)
  if (bid == 3 * B_) {  // plain f16 conversion of Wp
    f16* Wdst = Wfold + (size_t)24 * C_ * C_;
    for (int i = t; i < C_ * C_; i += 128) Wdst[i] = (f16)Wp[i];
    bfold[24 * C_ + t] = bp[t];
    return;
  }
  const int path = bid / B_, b = bid % B_;
  const float *g1, *be1, *g2 = nullptr, *be2 = nullptr, *Wsrc, *bsrc;
  if (path == 0)      { g1 = g_nq;  be1 = b_nq;  Wsrc = Wq; bsrc = bq; }
  else if (path == 1) { g1 = g_srk; be1 = b_srk; g2 = g_nk; be2 = b_nk; Wsrc = Wk; bsrc = bk; }
  else                { g1 = g_srv; be1 = b_srv; g2 = g_nv; be2 = b_nv; Wsrc = Wv; bsrc = bv; }

  const float npix = (path == 0) ? (float)LQ_ : (float)LK_;
  const float* st = stats + ((size_t)path * B_ * C_ + (size_t)b * C_) * 2;
  const float mu = st[t * 2] / npix, ex2 = st[t * 2 + 1] / npix;

  __shared__ float smu[128], se2[128], Dsh[128];
  smu[t] = mu; se2[t] = ex2;
  __syncthreads();
  const int g0 = t & ~3;  // 4 channels per group
  float gmu = 0.25f * (smu[g0] + smu[g0 + 1] + smu[g0 + 2] + smu[g0 + 3]);
  float ge2 = 0.25f * (se2[g0] + se2[g0 + 1] + se2[g0 + 2] + se2[g0 + 3]);
  float a1 = g1[t] * rsqrtf(ge2 - gmu * gmu + EPS_);
  float d1 = be1[t] - gmu * a1;
  float Ac, Dc;
  if (path == 0) {
    Ac = a1 * SCALE_; Dc = d1 * SCALE_;
  } else {
    float m1 = a1 * mu + d1;
    float e1 = a1 * a1 * ex2 + 2.f * a1 * d1 * mu + d1 * d1;
    __syncthreads();
    smu[t] = m1; se2[t] = e1;
    __syncthreads();
    float g2mu = 0.25f * (smu[g0] + smu[g0 + 1] + smu[g0 + 2] + smu[g0 + 3]);
    float g2e2 = 0.25f * (se2[g0] + se2[g0 + 1] + se2[g0 + 2] + se2[g0 + 3]);
    float a2 = g2[t] * rsqrtf(g2e2 - g2mu * g2mu + EPS_);
    float d2 = be2[t] - g2mu * a2;
    Ac = a2 * a1; Dc = a2 * d1 + d2;
  }
  Dsh[t] = Dc;
  __syncthreads();
  f16* Wdst = Wfold + (size_t)(path * B_ + b) * C_ * C_;
  for (int o = 0; o < C_; ++o) Wdst[o * C_ + t] = (f16)(Wsrc[o * C_ + t] * Ac);
  float bias = bsrc[t];
  if (path == 0) bias *= SCALE_;
  for (int c = 0; c < C_; ++c) bias += Wsrc[t * C_ + c] * Dsh[c];
  bfold[(path * B_ + b) * C_ + t] = bias;
}

// ---------------------------------------------------------------------------
// K3: per-batch GEMM  Y = W'(128x128 f16) @ X(128xLp) + b'.
// mode 0: X=q_src f32, out f16 [b][h][p][d]        (Lp=4096)
// mode 1: X=pooled_k,  out f16 [b][h][p][d]        (Lp=1024)
// mode 2: X=pooled_v,  out f16 [b][h][d][p] (T)    (Lp=1024)
// mode 3: X=oh f16 (head layout), out f32 NCHW to d_out (Lp=4096)
// grid (Lp/64, B), 128 threads = 4 waves; wave w -> N columns [w*16, w*16+16).
// ---------------------------------------------------------------------------
__global__ void conv_gemm_kernel(const float* __restrict__ Xf32,
                                 const f16* __restrict__ Xf16,
                                 const f16* __restrict__ Wfold,
                                 const float* __restrict__ bfold,
                                 f16* __restrict__ Yf16, float* __restrict__ Yf32,
                                 int Lp, int mode) {
  const int b  = blockIdx.y;
  const int p0 = blockIdx.x * 64;
  const int t = threadIdx.x, lane = t & 31, wv = t >> 5;
  const int slot = (mode == 3) ? 24 : (mode * B_ + b);
  const f16*   Wb = Wfold + (size_t)slot * C_ * C_;
  const float* bb = bfold + (size_t)slot * C_;

  // Pre-touch the W' panel this block will reuse across all 4 k-steps.
  __builtin_prefetch(Wb + (size_t)t * C_, 0, 1);  // -> global_prefetch_b8

  __shared__ alignas(16) f16 Xs[64][34];  // n-major (p-major) chunk: [p][c]
  v8f acc[8];
#pragma unroll
  for (int mt = 0; mt < 8; ++mt) acc[mt] = zero8();

  for (int k0 = 0; k0 < C_; k0 += 32) {
    __syncthreads();
    for (int e = t; e < 64 * 32; e += 128) {
      int p = e & 63, c = e >> 6;
      float x;
      if (mode == 3) {
        int ch = k0 + c;
        x = (float)Xf16[(((size_t)(b * HEADS_ + (ch >> 5))) * LQ_ + p0 + p) * DH_ + (ch & 31)];
      } else {
        x = Xf32[((size_t)b * C_ + k0 + c) * Lp + p0 + p];
      }
      Xs[p][c] = (f16)x;
    }
    __syncthreads();
    v16h bf = load_b16(&Xs[0][0], 34, wv * 16, 0, lane);
#pragma unroll
    for (int mt = 0; mt < 8; ++mt) {
      v16h af = load_a16(Wb, C_, mt * 16, k0, lane);
      acc[mt] = wmma_f16(af, bf, acc[mt]);
    }
  }
  // epilogue
  const int half = lane >> 4;
  const int p = p0 + wv * 16 + (lane & 15);
#pragma unroll
  for (int mt = 0; mt < 8; ++mt) {
#pragma unroll
    for (int r = 0; r < 8; ++r) {
      int o = mt * 16 + r + 8 * half;
      float val = acc[mt][r] + bb[o];
      int h = o >> 5, dd = o & 31;
      if (mode == 0 || mode == 1) {
        Yf16[(((size_t)(b * HEADS_ + h)) * Lp + p) * DH_ + dd] = (f16)val;
      } else if (mode == 2) {
        Yf16[(((size_t)(b * HEADS_ + h)) * DH_ + dd) * Lp + p] = (f16)val;
      } else {
        Yf32[((size_t)b * C_ + o) * LQ_ + p] = val;
      }
    }
  }
}

// ---------------------------------------------------------------------------
// K4: flash attention per (b, h).  Q pre-scaled by SCALE.
// grid (LQ/64, HEADS, B), 128 threads = 4 waves; wave owns a 16-row Q tile.
// K/V chunks staged with CDNA5 async global->LDS B128 copies (ASYNCcnt).
// ---------------------------------------------------------------------------
__global__ void attn_kernel(const f16* __restrict__ qh, const f16* __restrict__ kh,
                            const f16* __restrict__ vht, f16* __restrict__ oh) {
  const int b = blockIdx.z, h = blockIdx.y;
  const int t = threadIdx.x, lane = t & 31, wv = t >> 5;
  const int p0 = blockIdx.x * 64 + wv * 16;
  const f16* Q  = qh  + ((size_t)(b * HEADS_ + h)) * LQ_ * DH_;
  const f16* K  = kh  + ((size_t)(b * HEADS_ + h)) * LK_ * DH_;
  const f16* Vt = vht + ((size_t)(b * HEADS_ + h)) * DH_ * LK_;

  __shared__ alignas(16) f16 Ks[32][34];       // [j][d]   (n-major for QK^T B)
  __shared__ alignas(16) f16 Vs[32][34];       // [dd][j]  (n-major for PV  B)
  __shared__ alignas(16) f16 Ps[4][16][34];    // per-wave P, row-major [m][j]

  const v16h qf = load_a16(Q, DH_, p0, 0, lane);  // whole 16x32 Q tile
  v8f O0 = zero8(), O1 = zero8();
  float rm[8], rl[8];
#pragma unroll
  for (int r = 0; r < 8; ++r) { rm[r] = -1e30f; rl[r] = 0.f; }
  const int half = lane >> 4;
  const int jj = lane & 15;

  // Per-thread 16B transfer slots: row = t>>2, 16B chunk = t&3 (32 rows x 64B).
  const int srow = t >> 2, schunk = (t & 3) * 16;  // byte offset within 64B row
  const uint32_t lds_k = lds_addr32(&Ks[srow][0]) + schunk;
  const uint32_t lds_v = lds_addr32(&Vs[srow][0]) + schunk;

  for (int j0 = 0; j0 < LK_; j0 += 32) {
    __syncthreads();
    // Async-stage K chunk (32 x 32 f16, contiguous) and Vt chunk (rows at stride LK).
    async_copy_b128(lds_k, (uint64_t)(uintptr_t)(K + (size_t)j0 * DH_) + (uint64_t)t * 16);
    async_copy_b128(lds_v, (uint64_t)(uintptr_t)(Vt + (size_t)srow * LK_ + j0) + (uint64_t)schunk);
    wait_async0();
    __syncthreads();
    v8f S0 = zero8(), S1 = zero8();
    S0 = wmma_f16(qf, load_b16(&Ks[0][0], 34, 0, 0, lane), S0);
    S1 = wmma_f16(qf, load_b16(&Ks[0][0], 34, 16, 0, lane), S1);
    // online softmax across the 32 new columns (16-lane row reductions)
#pragma unroll
    for (int r = 0; r < 8; ++r) {
      float s0 = S0[r], s1 = S1[r];
      float mx = fmaxf(s0, s1);
#pragma unroll
      for (int off = 1; off < 16; off <<= 1) mx = fmaxf(mx, __shfl_xor(mx, off, 32));
      float rmn   = fmaxf(rm[r], mx);
      float alpha = __expf(rm[r] - rmn);
      float e0 = __expf(s0 - rmn), e1 = __expf(s1 - rmn);
      float ss = e0 + e1;
#pragma unroll
      for (int off = 1; off < 16; off <<= 1) ss += __shfl_xor(ss, off, 32);
      rl[r] = rl[r] * alpha + ss;
      rm[r] = rmn;
      O0[r] *= alpha; O1[r] *= alpha;
      int m = r + 8 * half;
      Ps[wv][m][jj]      = (f16)e0;
      Ps[wv][m][16 + jj] = (f16)e1;
    }
    v16h pa = load_a16(&Ps[wv][0][0], 34, 0, 0, lane);
    O0 = wmma_f16(pa, load_b16(&Vs[0][0], 34, 0, 0, lane), O0);
    O1 = wmma_f16(pa, load_b16(&Vs[0][0], 34, 16, 0, lane), O1);
  }
  f16* Ob = oh + ((size_t)(b * HEADS_ + h)) * LQ_ * DH_;
#pragma unroll
  for (int r = 0; r < 8; ++r) {
    int m = r + 8 * half;
    float inv = 1.0f / rl[r];
    Ob[((size_t)(p0 + m)) * DH_ + jj]      = (f16)(O0[r] * inv);
    Ob[((size_t)(p0 + m)) * DH_ + 16 + jj] = (f16)(O1[r] * inv);
  }
}

// ---------------------------------------------------------------------------
extern "C" void kernel_launch(void* const* d_in, const int* in_sizes, int n_in,
                              void* d_out, int out_size, void* d_ws, size_t ws_size,
                              hipStream_t stream) {
  const float* q_src = (const float*)d_in[0];
  const float* k_src = (const float*)d_in[1];
  const float* v_src = (const float*)d_in[2];
  const float* Wq = (const float*)d_in[3];  const float* bq = (const float*)d_in[4];
  const float* Wk = (const float*)d_in[5];  const float* bk = (const float*)d_in[6];
  const float* Wv = (const float*)d_in[7];  const float* bv = (const float*)d_in[8];
  const float* Wp = (const float*)d_in[9];  const float* bp = (const float*)d_in[10];
  const float* g_nq  = (const float*)d_in[11]; const float* b_nq  = (const float*)d_in[12];
  const float* g_nk  = (const float*)d_in[13]; const float* b_nk  = (const float*)d_in[14];
  const float* g_nv  = (const float*)d_in[15]; const float* b_nv  = (const float*)d_in[16];
  const float* g_srk = (const float*)d_in[17]; const float* b_srk = (const float*)d_in[18];
  const float* g_srv = (const float*)d_in[19]; const float* b_srv = (const float*)d_in[20];

  char* ws = (char*)d_ws;
  size_t off = 0;
  float* pooled_k = (float*)(ws + off); off += (size_t)B_ * C_ * LK_ * 4;       // 4 MB
  float* pooled_v = (float*)(ws + off); off += (size_t)B_ * C_ * LK_ * 4;       // 4 MB
  float* stats    = (float*)(ws + off); off += (size_t)3 * B_ * C_ * 2 * 4;     // 24 KB
  f16*   Wfold    = (f16*)(ws + off);   off += (size_t)25 * C_ * C_ * 2;        // 800 KB
  float* bfold    = (float*)(ws + off); off += (size_t)25 * C_ * 4;
  off = (off + 255) & ~(size_t)255;
  f16* qh  = (f16*)(ws + off); off += (size_t)B_ * HEADS_ * LQ_ * DH_ * 2;      // 8 MB
  f16* kh  = (f16*)(ws + off); off += (size_t)B_ * HEADS_ * LK_ * DH_ * 2;      // 2 MB
  f16* vht = (f16*)(ws + off); off += (size_t)B_ * HEADS_ * DH_ * LK_ * 2;      // 2 MB
  f16* oh  = (f16*)(ws + off); off += (size_t)B_ * HEADS_ * LQ_ * DH_ * 2;      // 8 MB

  pool_stats_kernel<<<3 * B_ * C_, 128, 0, stream>>>(q_src, k_src, v_src,
                                                     pooled_k, pooled_v, stats);
  fold_kernel<<<3 * B_ + 1, 128, 0, stream>>>(Wq, bq, Wk, bk, Wv, bv, Wp, bp,
                                              g_nq, b_nq, g_nk, b_nk, g_nv, b_nv,
                                              g_srk, b_srk, g_srv, b_srv,
                                              stats, Wfold, bfold);
  conv_gemm_kernel<<<dim3(LQ_ / 64, B_), 128, 0, stream>>>(q_src, nullptr, Wfold, bfold,
                                                           qh, nullptr, LQ_, 0);
  conv_gemm_kernel<<<dim3(LK_ / 64, B_), 128, 0, stream>>>(pooled_k, nullptr, Wfold, bfold,
                                                           kh, nullptr, LK_, 1);
  conv_gemm_kernel<<<dim3(LK_ / 64, B_), 128, 0, stream>>>(pooled_v, nullptr, Wfold, bfold,
                                                           vht, nullptr, LK_, 2);
  attn_kernel<<<dim3(LQ_ / 64, HEADS_, B_), 128, 0, stream>>>(qh, kh, vht, oh);
  conv_gemm_kernel<<<dim3(LQ_ / 64, B_), 128, 0, stream>>>(nullptr, oh, Wfold, bfold,
                                                           nullptr, (float*)d_out, LQ_, 3);
}
